// NeuralRDE_60069412602059
// MI455X (gfx1250) — compile-verified, matrix operand order checked by reading, not run on hardware
//
#include <hip/hip_runtime.h>
#include <hip/hip_bf16.h>

typedef __bf16 bf16;
typedef __attribute__((ext_vector_type(16))) __bf16 v16bf;
typedef __attribute__((ext_vector_type(8)))  float  v8f;

union FragBF { v16bf v; uint4 u[2]; };

#define NB 16      // batch
#define NT 512     // timesteps
#define ND 128     // state dim
#define NH 256     // hidden
#define NL 136     // logsig size
#define NLP 144    // padded logsig (9 tiles of 16)
#define NO 32      // output dim
#define NSTEP 511
#define NWG 32

#define LOG2E 1.4426950408889634f
#define LN2   0.6931471805599453f

// ---- fragment loader: row-major bf16 source, lane r = row(A)/col(B),
// two contiguous 8-elem K chunks per lane per the 16-bit 16x32 ISA layout.
static __device__ inline v16bf ldfrag(const bf16* base, int strideK, int kk, int lane) {
    int r  = lane & 15;
    int hi = (lane >> 4) << 3;           // 0 or 8
    const bf16* p = base + (size_t)r * strideK + kk * 32 + hi;
    FragBF f;
    f.u[0] = *(const uint4*)(p);         // K = kk*32+hi .. +7
    f.u[1] = *(const uint4*)(p + 16);    // K = kk*32+16+hi .. +7
    return f.v;
}

// raw hardware transcendentals (base-2); args are always in safe ranges here
static __device__ inline float hw_exp2(float x) {
#if __has_builtin(__builtin_amdgcn_exp2f)
    return __builtin_amdgcn_exp2f(x);
#else
    return __expf(x * LN2);
#endif
}
static __device__ inline float hw_log2(float x) {
#if __has_builtin(__builtin_amdgcn_logf)
    return __builtin_amdgcn_logf(x);
#else
    return __logf(x) * LOG2E;
#endif
}

// softplus(x) = max(x,0) + ln2 * log2(1 + 2^(-|x|*log2e)); log2 arg in (1,2]
static __device__ inline float softplus_f(float x) {
    float e = hw_exp2(-fabsf(x) * LOG2E);
    return fmaxf(x, 0.f) + LN2 * hw_log2(1.f + e);
}

// tanh: prefer gfx1250 hardware v_tanh_f32; else 1 - 2*rcp(2^(2x*log2e)+1)
static __device__ inline float fast_tanh(float x) {
#if __has_builtin(__builtin_amdgcn_tanhf)
    return __builtin_amdgcn_tanhf(x);
#else
    float e = hw_exp2(2.f * LOG2E * x);
    return 1.f - 2.f * __builtin_amdgcn_rcpf(e + 1.f);
#endif
}

// ---- sense-reversal grid barrier (32 resident WGs) ----
static __device__ inline void grid_barrier(unsigned* cnt, unsigned* gen) {
    __syncthreads();
    if (threadIdx.x == 0) {
        __threadfence();   // release all prior writes to device scope
        unsigned g = __hip_atomic_load(gen, __ATOMIC_RELAXED, __HIP_MEMORY_SCOPE_AGENT);
        unsigned a = __hip_atomic_fetch_add(cnt, 1u, __ATOMIC_ACQ_REL, __HIP_MEMORY_SCOPE_AGENT);
        if (a == NWG - 1) {
            __hip_atomic_store(cnt, 0u, __ATOMIC_RELAXED, __HIP_MEMORY_SCOPE_AGENT);
            __hip_atomic_store(gen, g + 1u, __ATOMIC_RELEASE, __HIP_MEMORY_SCOPE_AGENT);
        } else {
            while (__hip_atomic_load(gen, __ATOMIC_ACQUIRE, __HIP_MEMORY_SCOPE_AGENT) == g)
                __builtin_amdgcn_s_sleep(1);
        }
        __threadfence();   // acquire
    }
    __syncthreads();
}

// ================= prep: bf16 conversion + fw3 packing [128][144][256] =================
__global__ void k_prep(const float* __restrict__ fw1, const float* __restrict__ fw2,
                       const float* __restrict__ rw,  const float* __restrict__ fw3,
                       bf16* fw1b, bf16* fw2b, bf16* rwb, bf16* fw3p, unsigned* bar) {
    size_t i = (size_t)blockIdx.x * blockDim.x + threadIdx.x;
    size_t st = (size_t)gridDim.x * blockDim.x;
    if (i == 0) { bar[0] = 0u; bar[1] = 0u; }
    for (size_t j = i; j < (size_t)NH * ND;  j += st) fw1b[j] = (bf16)fw1[j];
    for (size_t j = i; j < (size_t)NH * NH;  j += st) fw2b[j] = (bf16)fw2[j];
    for (size_t j = i; j < (size_t)NO * ND;  j += st) rwb[j]  = (bf16)rw[j];
    size_t tot = (size_t)ND * NLP * NH;
    for (size_t j = i; j < tot; j += st) {
        size_t k   = j & 255;
        size_t row = j >> 8;
        size_t lp  = row % NLP;
        size_t d   = row / NLP;
        fw3p[j] = (lp < NL) ? (bf16)fw3[(d * NL + lp) * NH + k] : (bf16)0.f;
    }
}

// ================= initial MLP: x0 -> h0 (tiny, plain fp32) =================
__global__ void k_init(const float* __restrict__ x0,
                       const float* __restrict__ iw1, const float* __restrict__ ib1,
                       const float* __restrict__ iw2, const float* __restrict__ ib2,
                       const float* __restrict__ iw3, const float* __restrict__ ib3,
                       float* hhist, bf16* hhistb) {
    __shared__ float t1[NB * NH];
    __shared__ float t2[NB * NH];
    int tid = threadIdx.x;
    for (int i = tid; i < NB * NH; i += 256) {
        int b = i >> 8, n = i & 255;
        float acc = ib1[n];
        #pragma unroll
        for (int k = 0; k < 16; ++k) acc += x0[b * 16 + k] * iw1[n * 16 + k];
        t1[i] = softplus_f(acc);
    }
    __syncthreads();
    for (int i = tid; i < NB * NH; i += 256) {
        int b = i >> 8, n = i & 255;
        float acc = ib2[n];
        for (int k = 0; k < NH; ++k) acc += t1[b * NH + k] * iw2[n * NH + k];
        t2[i] = softplus_f(acc);
    }
    __syncthreads();
    for (int i = tid; i < NB * ND; i += 256) {
        int b = i >> 7, d = i & 127;
        float acc = ib3[d];
        for (int k = 0; k < NH; ++k) acc += t2[b * NH + k] * iw3[d * NH + k];
        hhist[i]  = acc;
        hhistb[i] = (bf16)acc;
    }
}

// ================= persistent scan kernel =================
__launch_bounds__(256, 1)
__global__ void k_scan(const float* __restrict__ logsigs,
                       const bf16* __restrict__ fw1b, const float* __restrict__ fb1,
                       const bf16* __restrict__ fw2b, const float* __restrict__ fb2,
                       const bf16* __restrict__ fw3p, const float* __restrict__ fb3,
                       float* hhist, bf16* hhistb, unsigned* bar) {
    __shared__ __align__(16) bf16 hbs[NB * ND];    // h_t in bf16
    __shared__ __align__(16) bf16 z1s[NB * NH];
    __shared__ __align__(16) bf16 z2s[NB * NH];
    __shared__ float sigs[NB * NLP];               // logsig_t, zero-padded to 144
    __shared__ float part[8][16];

    const int tid  = threadIdx.x;
    const int lane = tid & 31;
    const int w    = tid >> 5;                     // wave 0..7
    const int l15  = lane & 15;
    const int mb   = (lane >> 4) << 3;             // row base 0 or 8 for C tiles

    for (int t = 0; t < NSTEP; ++t) {
        // ---- stage h_t (bf16) and logsig_t into LDS ----
        const bf16* hb = hhistb + (size_t)t * (NB * ND);
        #pragma unroll
        for (int i = 0; i < 8; ++i) hbs[tid + 256 * i] = hb[tid + 256 * i];
        #pragma unroll
        for (int i = 0; i < 9; ++i) {
            int idx = tid + 256 * i;
            int b = idx / NLP, lp = idx - b * NLP;
            sigs[idx] = (lp < NL) ? logsigs[((size_t)b * NSTEP + t) * NL + lp] : 0.f;
        }
        __syncthreads();

        // ---- z1 = softplus(h @ fw1^T + fb1): N=256 -> 16 tiles / 8 waves ----
        #pragma unroll
        for (int rep = 0; rep < 2; ++rep) {
            int nt = w + rep * 8;
            v8f c = {0.f, 0.f, 0.f, 0.f, 0.f, 0.f, 0.f, 0.f};
            const bf16* wb = fw1b + (size_t)nt * 16 * ND;
            #pragma unroll
            for (int kk = 0; kk < 4; ++kk) {
                v16bf a = ldfrag(hbs, ND, kk, lane);
                v16bf b = ldfrag(wb, ND, kk, lane);
                c = __builtin_amdgcn_wmma_f32_16x16x32_bf16(false, a, false, b,
                                                            (short)0, c, false, false);
            }
            int n = nt * 16 + l15;
            float bias = fb1[n];
            #pragma unroll
            for (int v = 0; v < 8; ++v)
                z1s[(mb + v) * NH + n] = (bf16)softplus_f(c[v] + bias);
        }
        __syncthreads();

        // ---- z2 = softplus(z1 @ fw2^T + fb2) ----
        #pragma unroll
        for (int rep = 0; rep < 2; ++rep) {
            int nt = w + rep * 8;
            v8f c = {0.f, 0.f, 0.f, 0.f, 0.f, 0.f, 0.f, 0.f};
            const bf16* wb = fw2b + (size_t)nt * 16 * NH;
            #pragma unroll
            for (int kk = 0; kk < 8; ++kk) {
                v16bf a = ldfrag(z1s, NH, kk, lane);
                v16bf b = ldfrag(wb, NH, kk, lane);
                c = __builtin_amdgcn_wmma_f32_16x16x32_bf16(false, a, false, b,
                                                            (short)0, c, false, false);
            }
            int n = nt * 16 + l15;
            float bias = fb2[n];
            #pragma unroll
            for (int v = 0; v < 8; ++v)
                z2s[(mb + v) * NH + n] = (bf16)softplus_f(c[v] + bias);
        }
        __syncthreads();

        // ---- big GEMM slice + tanh + logsig contraction ----
        // wave pair (2j, 2j+1) owns d = blockIdx.x*4 + j; tiles 0..3 / 4..8 of 9.
        const int dl   = w >> 1;
        const int d    = blockIdx.x * 4 + dl;
        const int half = w & 1;
        const int tt0  = half ? 4 : 0;
        const int tt1  = half ? 9 : 4;

        v16bf a8[8];                                  // cache A fragments (z2)
        #pragma unroll
        for (int kk = 0; kk < 8; ++kk) a8[kk] = ldfrag(z2s, NH, kk, lane);

        float s[8];
        #pragma unroll
        for (int v = 0; v < 8; ++v) s[v] = 0.f;

        for (int tt = tt0; tt < tt1; ++tt) {
            v8f c = {0.f, 0.f, 0.f, 0.f, 0.f, 0.f, 0.f, 0.f};
            const bf16* bp = fw3p + ((size_t)d * NLP + tt * 16) * NH;
            // pull next B tile (8 KB) toward WGP while this tile computes
            if (tt + 1 < tt1)
                __builtin_prefetch(bp + 16 * NH + (size_t)l15 * NH, 0, 3);
            #pragma unroll
            for (int kk = 0; kk < 8; ++kk) {
                v16bf b = ldfrag(bp, NH, kk, lane);
                c = __builtin_amdgcn_wmma_f32_16x16x32_bf16(false, a8[kk], false, b,
                                                            (short)0, c, false, false);
            }
            int l = tt * 16 + l15;                    // padded-l column of this lane
            float bias = (l < NL) ? fb3[(size_t)d * NL + l] : 0.f;
            #pragma unroll
            for (int v = 0; v < 8; ++v) {
                float x = fast_tanh(c[v] + bias);
                s[v] += x * sigs[(mb + v) * NLP + l]; // sigs==0 on padded cols
            }
        }
        // reduce over the 16 lanes of each half (columns)
        #pragma unroll
        for (int m = 1; m < 16; m <<= 1) {
            #pragma unroll
            for (int v = 0; v < 8; ++v) s[v] += __shfl_xor(s[v], m, 32);
        }
        if (lane == 0) {
            #pragma unroll
            for (int v = 0; v < 8; ++v) part[w][v] = s[v];
        }
        if (lane == 16) {
            #pragma unroll
            for (int v = 0; v < 8; ++v) part[w][8 + v] = s[v];
        }
        __syncthreads();

        // ---- combine and write h_{t+1} (each (b,d) owned by exactly one WG) ----
        if (tid < 64) {
            int dl2 = tid >> 4, b = tid & 15;
            int dg  = blockIdx.x * 4 + dl2;
            float hn = hhist[(size_t)t * (NB * ND) + b * ND + dg]
                     + part[2 * dl2][b] + part[2 * dl2 + 1][b];
            hhist [(size_t)(t + 1) * (NB * ND) + b * ND + dg] = hn;
            hhistb[(size_t)(t + 1) * (NB * ND) + b * ND + dg] = (bf16)hn;
        }
        grid_barrier(bar, bar + 1);
    }
}

// ================= readout: out[b][t][o] = h_all[t][b] @ rw^T + rb =================
__global__ void k_readout(const bf16* __restrict__ hhistb, const bf16* __restrict__ rwb,
                          const float* __restrict__ rb, float* __restrict__ out) {
    const int tid  = threadIdx.x;
    const int lane = tid & 31;
    const int w    = tid >> 5;
    const int l15  = lane & 15;
    const int mb   = (lane >> 4) << 3;
    const int rt   = blockIdx.x * 8 + w;              // row tile == timestep t, 0..511
    const bf16* abase = hhistb + (size_t)rt * 16 * ND;
    #pragma unroll
    for (int nt = 0; nt < 2; ++nt) {
        v8f c = {0.f, 0.f, 0.f, 0.f, 0.f, 0.f, 0.f, 0.f};
        const bf16* wb = rwb + (size_t)nt * 16 * ND;
        #pragma unroll
        for (int kk = 0; kk < 4; ++kk) {
            v16bf a = ldfrag(abase, ND, kk, lane);
            v16bf b = ldfrag(wb, ND, kk, lane);
            c = __builtin_amdgcn_wmma_f32_16x16x32_bf16(false, a, false, b,
                                                        (short)0, c, false, false);
        }
        int o = nt * 16 + l15;
        float bias = rb[o];
        #pragma unroll
        for (int v = 0; v < 8; ++v) {
            int b = mb + v;                           // batch index
            out[((size_t)b * NT + rt) * NO + o] = c[v] + bias;
        }
    }
}

extern "C" void kernel_launch(void* const* d_in, const int* in_sizes, int n_in,
                              void* d_out, int out_size, void* d_ws, size_t ws_size,
                              hipStream_t stream) {
    const float* x0      = (const float*)d_in[0];
    const float* logsigs = (const float*)d_in[1];
    const float* iw1 = (const float*)d_in[2];
    const float* ib1 = (const float*)d_in[3];
    const float* iw2 = (const float*)d_in[4];
    const float* ib2 = (const float*)d_in[5];
    const float* iw3 = (const float*)d_in[6];
    const float* ib3 = (const float*)d_in[7];
    const float* fw1 = (const float*)d_in[8];
    const float* fb1 = (const float*)d_in[9];
    const float* fw2 = (const float*)d_in[10];
    const float* fb2 = (const float*)d_in[11];
    const float* fw3 = (const float*)d_in[12];
    const float* fb3 = (const float*)d_in[13];
    const float* rw  = (const float*)d_in[14];
    const float* rb  = (const float*)d_in[15];
    float* out = (float*)d_out;

    // workspace layout (256B-aligned regions)
    char* ws = (char*)d_ws;
    size_t off = 0;
    bf16* fw1b = (bf16*)(ws + off); off += (size_t)NH * ND * 2;            // 64 KB
    bf16* fw2b = (bf16*)(ws + off); off += (size_t)NH * NH * 2;            // 128 KB
    bf16* rwb  = (bf16*)(ws + off); off += (size_t)NO * ND * 2;            // 8 KB
    off = (off + 255) & ~(size_t)255;
    bf16* fw3p = (bf16*)(ws + off); off += (size_t)ND * NLP * NH * 2;      // 9.44 MB
    off = (off + 255) & ~(size_t)255;
    float* hhist = (float*)(ws + off); off += (size_t)NT * NB * ND * 4;    // 4 MB
    bf16* hhistb = (bf16*)(ws + off);  off += (size_t)NT * NB * ND * 2;    // 2 MB
    off = (off + 255) & ~(size_t)255;
    unsigned* bar = (unsigned*)(ws + off); off += 256;

    k_prep<<<2048, 256, 0, stream>>>(fw1, fw2, rw, fw3, fw1b, fw2b, rwb, fw3p, bar);
    k_init<<<1, 256, 0, stream>>>(x0, iw1, ib1, iw2, ib2, iw3, ib3, hhist, hhistb);
    k_scan<<<NWG, 256, 0, stream>>>(logsigs, fw1b, fb1, fw2b, fb2, fw3p, fb3,
                                    hhist, hhistb, bar);
    k_readout<<<64, 256, 0, stream>>>(hhistb, rwb, rb, out);
}
